// rgtn_b_1666447311037
// MI455X (gfx1250) — matrix operand order repaired and checked by reading, not compiled, not run on hardware
//
#include <hip/hip_runtime.h>
#include <stdint.h>

// ============================================================================
// RGTN forward for MI455X (gfx1250, wave32, WMMA).
//  - Dense projections: bf16 WMMA (v_wmma_f32_16x16x32_bf16), f32 accumulate,
//    LDS-tiled 128xNC per workgroup (8 waves). B weights are staged in
//    fragment order so every WMMA operand loads as 2x ds_load_b128; all B
//    fragments for a K-step are preloaded so WMMAs issue back-to-back.
//  - rel_emb@We collapsed to an 8x128 table (only 8 relation types).
//  - Edge attention: wave-per-edge (head dim 32 == wave32), L2 atomics.
// Input order assumption: setup_inputs() insertion order, params flattened
// in source order.
// ============================================================================

#define NN 20000
#define NE 320000
#define HD 128
#define RAN 9

typedef __attribute__((ext_vector_type(16))) __bf16 v16bf;
typedef __attribute__((ext_vector_type(8)))  float  v8f;

__device__ __forceinline__ unsigned short f2bf(float f) {
  unsigned u = __float_as_uint(f);
  u += 0x7fffu + ((u >> 16) & 1u);           // round-to-nearest-even
  return (unsigned short)(u >> 16);
}
// order-preserving float<->uint encoding for atomicMax on f32
__device__ __forceinline__ unsigned fenc(float f) {
  unsigned u = __float_as_uint(f);
  return (u & 0x80000000u) ? ~u : (u | 0x80000000u);
}
__device__ __forceinline__ float fdec(unsigned u) {
  unsigned v = (u & 0x80000000u) ? (u & 0x7fffffffu) : ~u;
  return __uint_as_float(v);
}

// ---------------------------------------------------------------------------
// staging converts
// ---------------------------------------------------------------------------
__global__ void cvt_pad_cols(const float* __restrict__ src, unsigned short* __restrict__ dst,
                             int M, int K, int Kp) {
  long total = (long)M * Kp;
  for (long i = blockIdx.x * (long)blockDim.x + threadIdx.x; i < total;
       i += (long)gridDim.x * blockDim.x) {
    long r = i / Kp; int c = (int)(i - r * Kp);
    float f = (c < K) ? src[r * K + c] : 0.0f;
    dst[i] = f2bf(f);
  }
}

// Stage weight W[K,NC] (f32, row-major) -> bf16 in WMMA B-fragment order:
//   dst[(((kblk*(NC/16))+c)*32 + lane)*16 + e]
//     = W[kblk*32 + (lane>>4)*16 + e][c*16 + (lane&15)]
// so lane L's 16 B elements for col-tile c are 32B-contiguous.
__global__ void cvt_swizzle_b(const float* __restrict__ src, unsigned short* __restrict__ dst,
                              int K, int Kp, int NC) {
  long total = (long)Kp * NC;
  int nct = NC >> 4;
  for (long i = blockIdx.x * (long)blockDim.x + threadIdx.x; i < total;
       i += (long)gridDim.x * blockDim.x) {
    int e = (int)(i & 15);
    long t = i >> 4;
    int lane = (int)(t & 31);
    long u = t >> 5;
    int c = (int)(u % nct);
    int kblk = (int)(u / nct);
    int k = kblk * 32 + ((lane >> 4) << 4) + e;
    int col = c * 16 + (lane & 15);
    float f = (k < K) ? src[(long)k * NC + col] : 0.0f;
    dst[i] = f2bf(f);
  }
}

__global__ void fill_f32(float* p, float v, long n) {
  for (long i = blockIdx.x * (long)blockDim.x + threadIdx.x; i < n;
       i += (long)gridDim.x * blockDim.x) p[i] = v;
}
__global__ void fill_u32(unsigned* p, unsigned v, long n) {
  for (long i = blockIdx.x * (long)blockDim.x + threadIdx.x; i < n;
       i += (long)gridDim.x * blockDim.x) p[i] = v;
}

// ---------------------------------------------------------------------------
// bf16 WMMA GEMM: C[M,NC] = A[M,Kp](bf16 row-major) x B(bf16, fragment order)
// 256 threads = 8 waves; block tile 128 x NC; wave w owns rows [16w,16w+16).
// ---------------------------------------------------------------------------
template <int NCT>
__global__ __launch_bounds__(256) void gemm_bf16_wmma(
    const unsigned short* __restrict__ A, int lda,
    const unsigned short* __restrict__ B,
    float* __restrict__ C, int M, int Kp,
    const float* __restrict__ bias, int act) {
  constexpr int NC = NCT * 16;
  __shared__ __align__(16) unsigned short sA[128 * 32];
  __shared__ __align__(16) unsigned short sB[32 * NC];
  const int tid = threadIdx.x;
  const int wv = tid >> 5;
  const int lane = tid & 31;
  const int rowBase = blockIdx.x * 128;
  const int m = lane & 15;
  const int hi = lane >> 4;

  v8f acc[NCT];
#pragma unroll
  for (int c = 0; c < NCT; c++)
#pragma unroll
    for (int r = 0; r < 8; r++) acc[c][r] = 0.0f;

  union Frag { v16bf v; uint4 u4[2]; };

  for (int kb = 0; kb < Kp; kb += 32) {
    {  // A tile 128x32 row-major, 16 ushorts (32B) per thread, uint4 moves
      int r = tid >> 1;
      int cs = (tid & 1) << 4;
      int grow = rowBase + r;
      uint4* d4 = reinterpret_cast<uint4*>(&sA[r * 32 + cs]);
      if (grow < M) {
        const uint4* s4 = reinterpret_cast<const uint4*>(A + (size_t)grow * lda + kb + cs);
        d4[0] = s4[0]; d4[1] = s4[1];
      } else {
        uint4 z; z.x = z.y = z.z = z.w = 0u;
        d4[0] = z; d4[1] = z;
      }
    }
    {  // B tile: already fragment-ordered in global -> identity copy
      int nthr = 2 * NC;  // (32*NC)/16 chunks of 16 ushorts
      if (tid < nthr) {
        const uint4* s4 =
            reinterpret_cast<const uint4*>(B + (size_t)(kb >> 5) * 32 * NC + (size_t)tid * 16);
        uint4* d4 = reinterpret_cast<uint4*>(&sB[tid * 16]);
        d4[0] = s4[0]; d4[1] = s4[1];
      }
    }
    __syncthreads();
    // A fragment, contiguous K order matching B staging: one 16-ushort run
    //   e0..15 -> K = hi*16 + e       (offset hi*16)
    Frag af;
    {
      const unsigned short* ap = &sA[(wv * 16 + m) * 32 + hi * 16];
      af.u4[0] = *reinterpret_cast<const uint4*>(ap);
      af.u4[1] = *reinterpret_cast<const uint4*>(ap + 8);
    }
    // Preload ALL B fragments (independent ds_load_b128 clause, one wait),
    // then issue the NCT WMMAs back-to-back (no D->A/B hazards between them).
    Frag bfr[NCT];
#pragma unroll
    for (int c = 0; c < NCT; c++) {
      const uint4* bp = reinterpret_cast<const uint4*>(&sB[(c * 32 + lane) * 16]);
      bfr[c].u4[0] = bp[0];
      bfr[c].u4[1] = bp[1];
    }
#pragma unroll
    for (int c = 0; c < NCT; c++) {
      acc[c] = __builtin_amdgcn_wmma_f32_16x16x32_bf16(
          false, af.v, false, bfr[c].v, (short)0, acc[c], false, false);
    }
    __syncthreads();
  }
  // C layout: VGPR r, lanes0-15 -> M=r, lanes16-31 -> M=r+8 ; N=lane%16
#pragma unroll
  for (int c = 0; c < NCT; c++) {
#pragma unroll
    for (int r = 0; r < 8; r++) {
      int row = rowBase + wv * 16 + (hi << 3) + r;
      if (row < M) {
        int col = c * 16 + m;
        float vv = acc[c][r];
        if (bias) vv += bias[col];
        if (act == 1) vv = vv > 0.0f ? vv : 0.0f;
        C[(size_t)row * NC + col] = vv;
      }
    }
  }
}

// ---------------------------------------------------------------------------
// rel_proj[8,128] = rel_emb[8,32] @ We[32,128]
// ---------------------------------------------------------------------------
__global__ void relproj_kernel(const float* __restrict__ rel_emb,
                               const float* __restrict__ We, float* __restrict__ out) {
  int j = blockIdx.x * blockDim.x + threadIdx.x;
  if (j < 8 * 128) {
    int r = j >> 7, c = j & 127;
    float s = 0.0f;
    for (int k = 0; k < 32; k++) s += rel_emb[r * 32 + k] * We[k * 128 + c];
    out[j] = s;
  }
}

// ---------------------------------------------------------------------------
// edge attention: wave per edge (head dim 32 == wave width)
// ---------------------------------------------------------------------------
__global__ void edge_score_kernel(const int* __restrict__ src, const int* __restrict__ dst,
                                  const int* __restrict__ et,
                                  const float* __restrict__ q, const float* __restrict__ k,
                                  const float* __restrict__ relp,
                                  float* __restrict__ score, unsigned* __restrict__ mmax, int E) {
  int e = blockIdx.x * (blockDim.x >> 5) + (threadIdx.x >> 5);
  int lane = threadIdx.x & 31;
  if (e >= E) return;
  int s = src[e], d = dst[e], t = et[e];
  const float* qd = q + (size_t)d * HD;
  const float* ks = k + (size_t)s * HD;
  const float* rp = relp + t * HD;
#pragma unroll
  for (int h = 0; h < 4; h++) {
    int idx = h * 32 + lane;
    float p = qd[idx] * (ks[idx] + rp[idx]);
    for (int off = 16; off > 0; off >>= 1) p += __shfl_down(p, off);
    if (lane == 0) {
      float sc = p * 0.17677669529663687f;  // 1/sqrt(32)
      score[(size_t)e * 4 + h] = sc;
      atomicMax(&mmax[(size_t)d * 4 + h], fenc(sc));
    }
  }
}

__global__ void edge_expw_kernel(const int* __restrict__ dst, const float* __restrict__ score,
                                 const unsigned* __restrict__ mmax, float* __restrict__ expw,
                                 float* __restrict__ zsum, long E4) {
  for (long i = blockIdx.x * (long)blockDim.x + threadIdx.x; i < E4;
       i += (long)gridDim.x * blockDim.x) {
    long e = i >> 2; int h = (int)(i & 3);
    int d = dst[e];
    float mx = fdec(mmax[(size_t)d * 4 + h]);
    float w = expf(score[i] - mx);
    expw[i] = w;
    atomicAdd(&zsum[(size_t)d * 4 + h], w);
  }
}

__global__ void edge_agg_kernel(const int* __restrict__ src, const int* __restrict__ dst,
                                const int* __restrict__ et,
                                const float* __restrict__ v, const float* __restrict__ relp,
                                const float* __restrict__ expw, const float* __restrict__ zsum,
                                float* __restrict__ attn, int E) {
  int e = blockIdx.x * (blockDim.x >> 5) + (threadIdx.x >> 5);
  int lane = threadIdx.x & 31;
  if (e >= E) return;
  int s = src[e], d = dst[e], t = et[e];
#pragma unroll
  for (int h = 0; h < 4; h++) {
    float w = expw[(size_t)e * 4 + h] / (zsum[(size_t)d * 4 + h] + 1e-9f);
    int idx = h * 32 + lane;
    atomicAdd(&attn[(size_t)d * HD + idx],
              w * (v[(size_t)s * HD + idx] + relp[t * HD + idx]));
  }
}

// ---------------------------------------------------------------------------
// out = LN(a+b)*g+be ; optionally also emit bf16 copy (next layer GEMM input)
// ---------------------------------------------------------------------------
__global__ void add_ln_kernel(const float* __restrict__ a, const float* __restrict__ b,
                              const float* __restrict__ g, const float* __restrict__ be,
                              float* __restrict__ out, unsigned short* __restrict__ obf, int n) {
  int row = blockIdx.x;
  if (row >= n) return;
  int j = threadIdx.x;
  __shared__ float red[128];
  float x = a[(size_t)row * HD + j] + b[(size_t)row * HD + j];
  red[j] = x; __syncthreads();
  for (int off = 64; off > 0; off >>= 1) { if (j < off) red[j] += red[j + off]; __syncthreads(); }
  float mean = red[0] * (1.0f / 128.0f);
  __syncthreads();
  float dx = x - mean;
  red[j] = dx * dx; __syncthreads();
  for (int off = 64; off > 0; off >>= 1) { if (j < off) red[j] += red[j + off]; __syncthreads(); }
  float var = red[0] * (1.0f / 128.0f);
  float y = dx * rsqrtf(var + 1e-6f) * g[j] + be[j];
  out[(size_t)row * HD + j] = y;
  if (obf) obf[(size_t)row * HD + j] = f2bf(y);
}

// q2 = LN(ffo+qo)*g+be + h2
__global__ void pff_ln_add_kernel(const float* __restrict__ ffo, const float* __restrict__ qo,
                                  const float* __restrict__ h2,
                                  const float* __restrict__ g, const float* __restrict__ be,
                                  float* __restrict__ q2, int rows) {
  int row = blockIdx.x;
  if (row >= rows) return;
  int j = threadIdx.x;
  __shared__ float red[128];
  float x = ffo[(size_t)row * HD + j] + qo[(size_t)row * HD + j];
  red[j] = x; __syncthreads();
  for (int off = 64; off > 0; off >>= 1) { if (j < off) red[j] += red[j + off]; __syncthreads(); }
  float mean = red[0] * (1.0f / 128.0f);
  __syncthreads();
  float dx = x - mean;
  red[j] = dx * dx; __syncthreads();
  for (int off = 64; off > 0; off >>= 1) { if (j < off) red[j] += red[j + off]; __syncthreads(); }
  float var = red[0] * (1.0f / 128.0f);
  float y = dx * rsqrtf(var + 1e-6f) * g[j] + be[j];
  q2[(size_t)row * HD + j] = y + h2[(size_t)row * HD + j];
}

// ---------------------------------------------------------------------------
// fusion stage
// ---------------------------------------------------------------------------
__global__ void stack_h2_kernel(const float* __restrict__ hs, const float* __restrict__ hc,
                                float* __restrict__ h2, unsigned short* __restrict__ h2bf,
                                long total) {
  for (long i = blockIdx.x * (long)blockDim.x + threadIdx.x; i < total;
       i += (long)gridDim.x * blockDim.x) {
    long nrow = i >> 8; int j = (int)(i & 255);
    float vv = (j < 128) ? hs[nrow * HD + j] : hc[nrow * HD + (j - 128)];
    h2[i] = vv;
    h2bf[i] = f2bf(vv);
  }
}

// 2x2 attention per node (wave per node)
__global__ void fuse_attn_kernel(const float* __restrict__ qf, const float* __restrict__ kf,
                                 const float* __restrict__ vf, float* __restrict__ qo,
                                 unsigned short* __restrict__ qobf, int n) {
  int node = blockIdx.x * (blockDim.x >> 5) + (threadIdx.x >> 5);
  int lane = threadIdx.x & 31;
  if (node >= n) return;
  const float* q = qf + (size_t)node * 256;
  const float* k = kf + (size_t)node * 256;
  const float* v = vf + (size_t)node * 256;
  float s[2][2];
#pragma unroll
  for (int i = 0; i < 2; i++)
#pragma unroll
    for (int j = 0; j < 2; j++) {
      float p = 0.0f;
      for (int d = lane; d < 128; d += 32) p += q[i * 128 + d] * k[j * 128 + d];
      for (int off = 16; off > 0; off >>= 1) p += __shfl_down(p, off);
      p = __shfl(p, 0);
      s[i][j] = p * 0.08838834764831845f;  // 1/sqrt(128)
    }
#pragma unroll
  for (int i = 0; i < 2; i++) {
    float mx = fmaxf(s[i][0], s[i][1]);
    float e0 = expf(s[i][0] - mx), e1 = expf(s[i][1] - mx);
    float z = e0 + e1;
    float a0 = e0 / z, a1 = e1 / z;
    for (int d = lane; d < 128; d += 32) {
      float o = a0 * v[d] + a1 * v[128 + d];
      qo[(size_t)node * 256 + i * 128 + d] = o;
      qobf[(size_t)node * 256 + i * 128 + d] = f2bf(o);
    }
  }
}

// batch-norm stats (per channel of [N,2,128])
__global__ void bn_stats_kernel(const float* __restrict__ q2, float* __restrict__ sums,
                                float* __restrict__ sqs, int n) {
  int c = threadIdx.x;  // 0..255
  float s = 0.0f, s2 = 0.0f;
  for (int row = blockIdx.x; row < n; row += gridDim.x) {
    float x = q2[(size_t)row * 256 + c];
    s += x; s2 += x * x;
  }
  atomicAdd(&sums[c], s);
  atomicAdd(&sqs[c], s2);
}

__global__ void bn_apply_kernel(float* __restrict__ q2, const float* __restrict__ sums,
                                const float* __restrict__ sqs,
                                const float* __restrict__ gs, const float* __restrict__ bs,
                                const float* __restrict__ gc, const float* __restrict__ bc,
                                int n) {
  long total = (long)n * 256;
  float inv = 1.0f / (float)n;
  for (long i = blockIdx.x * (long)blockDim.x + threadIdx.x; i < total;
       i += (long)gridDim.x * blockDim.x) {
    int c = (int)(i & 255);
    float mean = sums[c] * inv;
    float var = sqs[c] * inv - mean * mean;
    int br = c >> 7, ch = c & 127;
    float g = br ? gc[ch] : gs[ch];
    float b = br ? bc[ch] : bs[ch];
    q2[i] = (q2[i] - mean) * rsqrtf(var + 1e-5f) * g + b;
  }
}

// final head: per node (wave per node)
__global__ void head_kernel(const float* __restrict__ q2, const float* __restrict__ av,
                            const float* __restrict__ Wo1, const float* __restrict__ bo1,
                            const float* __restrict__ Wo2, const float* __restrict__ bo2,
                            float* __restrict__ logits, float* __restrict__ lsa,
                            float* __restrict__ lca, int n) {
  int node = blockIdx.x * (blockDim.x >> 5) + (threadIdx.x >> 5);
  int lane = threadIdx.x & 31;
  if (node >= n) return;
  const float* r0 = q2 + (size_t)node * 256;
  const float* r1 = r0 + 128;
  float a0 = 0, a1 = 0, u0 = 0, u1 = 0;
  for (int d = lane; d < 128; d += 32) {
    float a = av[d];
    a0 += r0[d] * a; a1 += r1[d] * a;
    u0 += r0[d] * Wo1[d]; u1 += r1[d] * Wo2[d];
  }
  for (int off = 16; off > 0; off >>= 1) {
    a0 += __shfl_down(a0, off); a1 += __shfl_down(a1, off);
    u0 += __shfl_down(u0, off); u1 += __shfl_down(u1, off);
  }
  if (lane == 0) {
    float ls = u0 + bo1[0]; ls = ls > 0.0f ? ls : 0.01f * ls;
    float lc = u1 + bo2[0]; lc = lc > 0.0f ? lc : 0.01f * lc;
    float mx = fmaxf(a0, a1);
    float e0 = expf(a0 - mx), e1 = expf(a1 - mx);
    float w0 = e0 / (e0 + e1), w1 = e1 / (e0 + e1);
    logits[node] = ls * w0 + lc * w1;
    lsa[node] = ls; lca[node] = lc;
  }
}

// ---------------------------------------------------------------------------
// losses
// ---------------------------------------------------------------------------
__global__ void mse_partial_kernel(const float* __restrict__ logits, const float* __restrict__ ls,
                                   const float* __restrict__ lc, const float* __restrict__ lab,
                                   float* __restrict__ acc, int n) {
  float p0 = 0, p1 = 0, p2 = 0;
  for (int i = blockIdx.x * blockDim.x + threadIdx.x; i < n; i += gridDim.x * blockDim.x) {
    float l = lab[i];
    float d0 = logits[i] - l, d1 = ls[i] - l, d2 = lc[i] - l;
    p0 += d0 * d0; p1 += d1 * d1; p2 += d2 * d2;
  }
  for (int off = 16; off > 0; off >>= 1) {
    p0 += __shfl_down(p0, off); p1 += __shfl_down(p1, off); p2 += __shfl_down(p2, off);
  }
  if ((threadIdx.x & 31) == 0) {
    atomicAdd(&acc[0], p0); atomicAdd(&acc[1], p1); atomicAdd(&acc[2], p2);
  }
}

__global__ void list_loss_kernel(const float* __restrict__ logits, const float* __restrict__ lab,
                                 const int* __restrict__ idx, float* __restrict__ acc, int n) {
  float part = 0.0f;
  for (int node = blockIdx.x * blockDim.x + threadIdx.x; node < n;
       node += gridDim.x * blockDim.x) {
    float lp[10], lt[10];
    lp[0] = logits[node]; lt[0] = lab[node];
    for (int j = 0; j < RAN; j++) {
      int m = idx[node * RAN + j];
      lp[j + 1] = logits[m]; lt[j + 1] = lab[m];
    }
    float mp = lp[0], mt = lt[0];
    for (int j = 1; j < 10; j++) { mp = fmaxf(mp, lp[j]); mt = fmaxf(mt, lt[j]); }
    float zp = 0, zt = 0;
    for (int j = 0; j < 10; j++) {
      lp[j] = expf(lp[j] - mp); zp += lp[j];
      lt[j] = expf(lt[j] - mt); zt += lt[j];
    }
    float s = 0.0f;
    for (int j = 0; j < 10; j++) s -= (lt[j] / zt) * logf(lp[j] / zp + 1e-10f);
    part += s;
  }
  for (int off = 16; off > 0; off >>= 1) part += __shfl_down(part, off);
  if ((threadIdx.x & 31) == 0) atomicAdd(&acc[3], part);
}

__global__ void finalize_kernel(const float* __restrict__ acc, float* __restrict__ out_loss,
                                int n) {
  if (blockIdx.x == 0 && threadIdx.x == 0) {
    float inv = 1.0f / (float)n;
    float mse0 = acc[0] * inv, mse1 = acc[1] * inv, mse2 = acc[2] * inv, ll = acc[3] * inv;
    out_loss[0] = 0.5f * mse0 + 0.5f * 0.5f * (mse1 + mse2) + 1.0f * ll;
  }
}

// ---------------------------------------------------------------------------
// host
// ---------------------------------------------------------------------------
static inline unsigned cdivu(long a, long b) { return (unsigned)((a + b - 1) / b); }

enum {
  IN_SRC = 0, IN_DST, IN_ETYPE, IN_STRUCT, IN_CONTENT, IN_LABELS, IN_RANK,
  P_REL_EMB,                 // 7
  P_SL0 = 8,                 // Wq,Wk,Wv,We,Wres,ln_g,ln_b
  P_SL1 = 15,
  P_CL0 = 22,
  P_CL1 = 29,
  P_WQF = 36, P_WKF = 37, P_WVF = 38,
  P_PFF_W1 = 39, P_PFF_B1 = 40, P_PFF_W2 = 41, P_PFF_B2 = 42,
  P_PFF_LN_G = 43, P_PFF_LN_B = 44,
  P_BN_S_G = 45, P_BN_S_B = 46, P_BN_C_G = 47, P_BN_C_B = 48,
  P_ATTN_VEC = 49,
  P_WO1 = 50, P_BO1 = 51, P_WO2 = 52, P_BO2 = 53
};

extern "C" void kernel_launch(void* const* d_in, const int* in_sizes, int n_in,
                              void* d_out, int out_size, void* d_ws, size_t ws_size,
                              hipStream_t stream) {
  (void)in_sizes; (void)n_in; (void)out_size; (void)ws_size;
  const int N = NN, E = NE;

  const int* srcI = (const int*)d_in[IN_SRC];
  const int* dstI = (const int*)d_in[IN_DST];
  const int* etI  = (const int*)d_in[IN_ETYPE];
  const float* structIn  = (const float*)d_in[IN_STRUCT];
  const float* contentIn = (const float*)d_in[IN_CONTENT];
  const float* labels    = (const float*)d_in[IN_LABELS];
  const int* rankIdx     = (const int*)d_in[IN_RANK];
  auto P = [&](int i) { return (const float*)d_in[i]; };

  // ---- workspace bump allocator (256B aligned) ----
  char* w = (char*)d_ws;
  auto alloc = [&](size_t bytes) {
    char* p = w;
    w += (bytes + 255) & ~(size_t)255;
    return p;
  };
  unsigned short* xbf = (unsigned short*)alloc((size_t)N * 320 * 2);  // node feat staging (bf16)
  unsigned short* wbf = (unsigned short*)alloc((size_t)320 * 128 * 2);// weight staging (bf16)
  float* hs = (float*)alloc((size_t)N * HD * 4);
  float* hc = (float*)alloc((size_t)N * HD * 4);
  const size_t POOL = (size_t)5 * 2 * N * HD * 4 + (512 + 2 * N + 64) * 4;
  char* pool = alloc(POOL);

  // graph-phase views of pool
  float* q_     = (float*)pool;
  float* k_     = q_ + (size_t)N * HD;
  float* v_     = k_ + (size_t)N * HD;
  float* resid_ = v_ + (size_t)N * HD;
  float* attn_  = resid_ + (size_t)N * HD;
  float* score_ = attn_ + (size_t)N * HD;
  float* expw_  = score_ + (size_t)E * 4;
  unsigned* mmax_ = (unsigned*)(expw_ + (size_t)E * 4);
  float* zsum_  = (float*)(mmax_ + (size_t)N * 4);
  float* relp_  = zsum_ + (size_t)N * 4;
  // fusion-phase views (alias pool; graph scratch dead by then)
  float* h2 = (float*)pool;
  float* qf = h2 + (size_t)2 * N * HD;
  float* kf = qf + (size_t)2 * N * HD;
  float* vf = kf + (size_t)2 * N * HD;
  float* qo = vf + (size_t)2 * N * HD;
  float* ff1 = vf;  // alias: vf dead after fuse_attn
  float* ffo = qf;  // alias: qf dead after fuse_attn
  float* q2  = kf;  // alias: kf dead after fuse_attn
  float* tail = qo + (size_t)2 * N * HD;
  float* bnsum = tail;
  float* bnsq  = bnsum + 256;
  float* lsa   = bnsq + 256;
  float* lca   = lsa + N;
  float* acc   = lca + N;

  float* outLogits = (float*)d_out;

  // ---- helpers ----
  auto gemm128 = [&](const float* W, int Kin, int Kp, float* C, int Mrows,
                     const float* bias, int act) {
    cvt_swizzle_b<<<cdivu((long)Kp * 128, 256), 256, 0, stream>>>(W, wbf, Kin, Kp, 128);
    gemm_bf16_wmma<8><<<cdivu(Mrows, 128), 256, 0, stream>>>(xbf, Kp, wbf, C, Mrows, Kp, bias, act);
  };
  auto gemm64 = [&](const float* W, int Kin, int Kp, float* C, int Mrows,
                    const float* bias, int act) {
    cvt_swizzle_b<<<cdivu((long)Kp * 64, 256), 256, 0, stream>>>(W, wbf, Kin, Kp, 64);
    gemm_bf16_wmma<4><<<cdivu(Mrows, 128), 256, 0, stream>>>(xbf, Kp, wbf, C, Mrows, Kp, bias, act);
  };

  auto run_layer = [&](const float* X, int Kin, int base, float* hout,
                       bool stageInput, bool emitNextBf) {
    int Kp = (Kin + 31) & ~31;
    if (stageInput)
      cvt_pad_cols<<<cdivu((long)N * Kp, 256), 256, 0, stream>>>(X, xbf, N, Kin, Kp);
    gemm128(P(base + 0), Kin, Kp, q_, N, nullptr, 0);     // Wq
    gemm128(P(base + 1), Kin, Kp, k_, N, nullptr, 0);     // Wk
    gemm128(P(base + 2), Kin, Kp, v_, N, nullptr, 0);     // Wv
    gemm128(P(base + 4), Kin, Kp, resid_, N, nullptr, 0); // Wres
    relproj_kernel<<<4, 256, 0, stream>>>(P(P_REL_EMB), P(base + 3), relp_);
    fill_f32<<<cdivu((long)N * HD, 256), 256, 0, stream>>>(attn_, 0.0f, (long)N * HD);
    fill_f32<<<cdivu((long)N * 4, 256), 256, 0, stream>>>(zsum_, 0.0f, (long)N * 4);
    fill_u32<<<cdivu((long)N * 4, 256), 256, 0, stream>>>(mmax_, 0u, (long)N * 4);
    edge_score_kernel<<<cdivu(E, 8), 256, 0, stream>>>(srcI, dstI, etI, q_, k_, relp_,
                                                       score_, mmax_, E);
    edge_expw_kernel<<<cdivu((long)E * 4, 256), 256, 0, stream>>>(dstI, score_, mmax_,
                                                                  expw_, zsum_, (long)E * 4);
    edge_agg_kernel<<<cdivu(E, 8), 256, 0, stream>>>(srcI, dstI, etI, v_, relp_,
                                                     expw_, zsum_, attn_, E);
    add_ln_kernel<<<N, 128, 0, stream>>>(attn_, resid_, P(base + 5), P(base + 6),
                                         hout, emitNextBf ? xbf : nullptr, N);
  };

  // ---- graph transformer stacks ----
  run_layer(structIn, 128, P_SL0, hs, true, true);
  run_layer(nullptr, 128, P_SL1, hs, false, false);
  run_layer(contentIn, 300, P_CL0, hc, true, true);
  run_layer(nullptr, 128, P_CL1, hc, false, false);

  // ---- fusion: h2 = stack([hs,hc]) as [2N,128] ----
  stack_h2_kernel<<<cdivu((long)N * 256, 256), 256, 0, stream>>>(hs, hc, h2, xbf,
                                                                 (long)N * 256);
  gemm128(P(P_WQF), 128, 128, qf, 2 * N, nullptr, 0);
  gemm128(P(P_WKF), 128, 128, kf, 2 * N, nullptr, 0);
  gemm128(P(P_WVF), 128, 128, vf, 2 * N, nullptr, 0);
  fuse_attn_kernel<<<cdivu(N, 8), 256, 0, stream>>>(qf, kf, vf, qo, xbf, N);
  // PFF: relu(qo@W1+b1)@W2+b2
  gemm64(P(P_PFF_W1), 128, 128, ff1, 2 * N, P(P_PFF_B1), 1);
  cvt_pad_cols<<<cdivu((long)2 * N * 64, 256), 256, 0, stream>>>(ff1, xbf, 2 * N, 64, 64);
  gemm128(P(P_PFF_W2), 64, 64, ffo, 2 * N, P(P_PFF_B2), 0);
  pff_ln_add_kernel<<<2 * N, 128, 0, stream>>>(ffo, qo, h2, P(P_PFF_LN_G), P(P_PFF_LN_B),
                                               q2, 2 * N);
  // batch norm (training-mode batch stats over node axis)
  fill_f32<<<2, 256, 0, stream>>>(bnsum, 0.0f, 512);
  bn_stats_kernel<<<240, 256, 0, stream>>>(q2, bnsum, bnsq, N);
  bn_apply_kernel<<<cdivu((long)N * 256, 256), 256, 0, stream>>>(
      q2, bnsum, bnsq, P(P_BN_S_G), P(P_BN_S_B), P(P_BN_C_G), P(P_BN_C_B), N);
  // head + losses
  head_kernel<<<cdivu(N, 8), 256, 0, stream>>>(q2, P(P_ATTN_VEC), P(P_WO1), P(P_BO1),
                                               P(P_WO2), P(P_BO2), outLogits, lsa, lca, N);
  fill_f32<<<1, 256, 0, stream>>>(acc, 0.0f, 4);
  mse_partial_kernel<<<120, 256, 0, stream>>>(outLogits, lsa, lca, labels, acc, N);
  list_loss_kernel<<<cdivu(N, 256), 256, 0, stream>>>(outLogits, labels, rankIdx, acc, N);
  finalize_kernel<<<1, 32, 0, stream>>>(acc, outLogits + N, N);
}
